// PLoRA_32607391711259
// MI455X (gfx1250) — matrix operand order repaired and checked by reading, not compiled, not run on hardware
//
#include <hip/hip_runtime.h>

// ---------------- CDNA5 WMMA vector types ----------------
typedef __attribute__((ext_vector_type(16))) __bf16 v16bf;
typedef __attribute__((ext_vector_type(8)))  __bf16 v8bf;
typedef __attribute__((ext_vector_type(4)))  __bf16 v4bf;
typedef __attribute__((ext_vector_type(8)))  float  v8f;

// Problem constants (from reference: B=4, S=2048, D_in=D_out=4096, R=8)
constexpr int M_TOT = 4 * 2048;   // 8192 tokens
constexpr int K_DIM = 4096;       // D_in
constexpr int N_DIM = 4096;       // D_out
constexpr int R_DIM = 8;          // LoRA rank
constexpr float LORA_SCALE = 16.0f / 8.0f;

// GEMM tiling
constexpr int BM = 128, BN = 128, BK = 32;
constexpr int LDT = 40;           // LDS row stride in bf16 (32 + 8 pad -> 80B, conflict-free b128 reads)

// Split f into hi bf16 + residual lo bf16 and write into vector lanes
#define SPLIT_STORE(val, j, hv, lv)              \
  {                                              \
    const float  _f = (val);                     \
    const __bf16 _h = (__bf16)_f;                \
    hv[j] = _h;                                  \
    lv[j] = (__bf16)(_f - (float)_h);            \
  }

// A fragment (16x32 bf16, ISA layout): lane<16 -> K{kb..kb+7, kb+16..kb+23}, kb=0; lane>=16 -> kb=8
__device__ __forceinline__ v16bf frag_a(const __bf16* rowp, int kb) {
  v8bf a = *(const v8bf*)(rowp + kb);
  v8bf b = *(const v8bf*)(rowp + kb + 16);
  return __builtin_shufflevector(a, b, 0, 1, 2, 3, 4, 5, 6, 7, 8, 9, 10, 11, 12, 13, 14, 15);
}

// B fragment (32x16 bf16, ISA layout): lane<16 -> col=lane, K{kb..kb+7, kb+16..kb+23}, kb=0; lane>=16 -> kb=8
__device__ __forceinline__ v16bf frag_b(const __bf16* colp, int kb) {
  v8bf a = *(const v8bf*)(colp + kb);
  v8bf b = *(const v8bf*)(colp + kb + 16);
  return __builtin_shufflevector(a, b, 0, 1, 2, 3, 4, 5, 6, 7, 8, 9, 10, 11, 12, 13, 14, 15);
}

// ---------------- Kernel 1: T = x @ W_A^T  (M_TOT x 8, fp32) ----------------
__global__ __launch_bounds__(256) void plora_lora_a_kernel(
    const float* __restrict__ x, const float* __restrict__ WA, float* __restrict__ T) {
  const int m   = blockIdx.x;
  const int tid = threadIdx.x;
  const float* xr = x + (size_t)m * K_DIM;

  float s[R_DIM];
#pragma unroll
  for (int r = 0; r < R_DIM; ++r) s[r] = 0.f;

  for (int d = tid * 4; d < K_DIM; d += 256 * 4) {
    const float4 xv = *(const float4*)(xr + d);
#pragma unroll
    for (int r = 0; r < R_DIM; ++r) {
      const float4 wv = *(const float4*)(WA + (size_t)r * K_DIM + d);
      s[r] = fmaf(xv.x, wv.x, s[r]);
      s[r] = fmaf(xv.y, wv.y, s[r]);
      s[r] = fmaf(xv.z, wv.z, s[r]);
      s[r] = fmaf(xv.w, wv.w, s[r]);
    }
  }
  // wave32 reduction
#pragma unroll
  for (int r = 0; r < R_DIM; ++r)
#pragma unroll
    for (int off = 16; off > 0; off >>= 1)
      s[r] += __shfl_xor(s[r], off, 32);

  __shared__ float red[8][R_DIM];
  const int wave = tid >> 5, lane = tid & 31;
  if (lane == 0) {
#pragma unroll
    for (int r = 0; r < R_DIM; ++r) red[wave][r] = s[r];
  }
  __syncthreads();
  if (tid < R_DIM) {
    float a = 0.f;
#pragma unroll
    for (int w = 0; w < 8; ++w) a += red[w][tid];
    T[(size_t)m * R_DIM + tid] = a;
  }
}

// ---------------- Kernel 2: fused main GEMM (bf16x3 WMMA) + bias + masked LoRA ----------------
__global__ __launch_bounds__(256) void plora_gemm_kernel(
    const float* __restrict__ x, const float* __restrict__ Wm,
    const float* __restrict__ bm, const float* __restrict__ WB,
    const float* __restrict__ T, const int* __restrict__ mask,
    float* __restrict__ out) {
  __shared__ __bf16 As_hi[BM * LDT];
  __shared__ __bf16 As_lo[BM * LDT];
  __shared__ __bf16 Bs_hi[BN * LDT];
  __shared__ __bf16 Bs_lo[BN * LDT];
  __shared__ float  Tl[BM * R_DIM];
  __shared__ int    Ml[BM];

  const int tid = threadIdx.x;
  const int n0  = blockIdx.x * BN;
  const int m0  = blockIdx.y * BM;

  // Stage LoRA intermediate + token mask for this 128-row panel
  *(float4*)(&Tl[tid * 4]) = *(const float4*)(T + (size_t)m0 * R_DIM + tid * 4);
  if (tid < BM) Ml[tid] = mask[m0 + tid];

  const int lane = tid & 31;
  const int wave = tid >> 5;
  const int wm   = wave & 3;   // 4 wave rows  -> 32 M each
  const int wn   = wave >> 2;  // 2 wave cols  -> 64 N each
  const int l15  = lane & 15;
  const int lhi  = lane >> 4;
  const int akb  = lhi ? 8  : 0;  // A K-slice select per ISA A-layout
  const int bkb  = lhi ? 8  : 0;  // B K-slice select per ISA B-layout

  // Per-thread staging coordinates (1024 float4 slots = 128 rows x 8 chunks)
  int srow[4], skc[4];
#pragma unroll
  for (int it = 0; it < 4; ++it) {
    const int idx = it * 256 + tid;
    srow[it] = idx >> 3;
    skc[it]  = (idx & 7) << 2;
  }

  const v8f vzero = {0.f, 0.f, 0.f, 0.f, 0.f, 0.f, 0.f, 0.f};
  v8f acc[2][4];
#pragma unroll
  for (int mi = 0; mi < 2; ++mi)
#pragma unroll
    for (int ni = 0; ni < 4; ++ni) acc[mi][ni] = vzero;

  const float* xp = x  + (size_t)m0 * K_DIM;
  const float* wp = Wm + (size_t)n0 * K_DIM;

  // Software pipeline: global float4s for the *next* K-tile live in registers
  float4 xv[4], wv[4];
#pragma unroll
  for (int it = 0; it < 4; ++it) {
    xv[it] = *(const float4*)(xp + (size_t)srow[it] * K_DIM + skc[it]);
    wv[it] = *(const float4*)(wp + (size_t)srow[it] * K_DIM + skc[it]);
  }

  for (int kt = 0; kt < K_DIM; kt += BK) {
    // ---- convert regs -> (hi, lo) bf16 LDS tiles ----
#pragma unroll
    for (int it = 0; it < 4; ++it) {
      v4bf ah4, al4, bh4, bl4;
      SPLIT_STORE(xv[it].x, 0, ah4, al4);
      SPLIT_STORE(xv[it].y, 1, ah4, al4);
      SPLIT_STORE(xv[it].z, 2, ah4, al4);
      SPLIT_STORE(xv[it].w, 3, ah4, al4);
      SPLIT_STORE(wv[it].x, 0, bh4, bl4);
      SPLIT_STORE(wv[it].y, 1, bh4, bl4);
      SPLIT_STORE(wv[it].z, 2, bh4, bl4);
      SPLIT_STORE(wv[it].w, 3, bh4, bl4);
      const int soff = srow[it] * LDT + skc[it];
      *(v4bf*)(&As_hi[soff]) = ah4;
      *(v4bf*)(&As_lo[soff]) = al4;
      *(v4bf*)(&Bs_hi[soff]) = bh4;
      *(v4bf*)(&Bs_lo[soff]) = bl4;
    }
    __syncthreads();

    // ---- issue next K-tile's global loads; they overlap the WMMAs below ----
    if (kt + BK < K_DIM) {
#pragma unroll
      for (int it = 0; it < 4; ++it) {
        const size_t go = (size_t)srow[it] * K_DIM + kt + BK + skc[it];
        xv[it] = *(const float4*)(xp + go);
        wv[it] = *(const float4*)(wp + go);
        // speculative prefetch two tiles ahead (silently dropped past the end)
        __builtin_prefetch(xp + go + BK, 0, 1);
        __builtin_prefetch(wp + go + BK, 0, 1);
      }
    }

    // ---- load fragments, 24 WMMAs per K-step per wave ----
    v16bf ah[2], al[2], bh[4], bl[4];
#pragma unroll
    for (int mi = 0; mi < 2; ++mi) {
      const int off = (wm * 32 + mi * 16 + l15) * LDT;
      ah[mi] = frag_a(As_hi + off, akb);
      al[mi] = frag_a(As_lo + off, akb);
    }
#pragma unroll
    for (int ni = 0; ni < 4; ++ni) {
      const int off = (wn * 64 + ni * 16 + l15) * LDT;
      bh[ni] = frag_b(Bs_hi + off, bkb);
      bl[ni] = frag_b(Bs_lo + off, bkb);
    }
#pragma unroll
    for (int mi = 0; mi < 2; ++mi)
#pragma unroll
      for (int ni = 0; ni < 4; ++ni) {
        acc[mi][ni] = __builtin_amdgcn_wmma_f32_16x16x32_bf16(
            false, ah[mi], false, bh[ni], (short)0, acc[mi][ni], false, false);
        acc[mi][ni] = __builtin_amdgcn_wmma_f32_16x16x32_bf16(
            false, ah[mi], false, bl[ni], (short)0, acc[mi][ni], false, false);
        acc[mi][ni] = __builtin_amdgcn_wmma_f32_16x16x32_bf16(
            false, al[mi], false, bh[ni], (short)0, acc[mi][ni], false, false);
      }
    __syncthreads();
  }

  // ---- epilogue: bias + masked LoRA (exact fp32 VALU), C layout per ISA ----
  float bias[4];
  float wbv[4][R_DIM];
#pragma unroll
  for (int ni = 0; ni < 4; ++ni) {
    const int n = n0 + wn * 64 + ni * 16 + l15;
    bias[ni] = bm[n];
    const float4 w0 = *(const float4*)(WB + (size_t)n * R_DIM);
    const float4 w1 = *(const float4*)(WB + (size_t)n * R_DIM + 4);
    wbv[ni][0] = w0.x; wbv[ni][1] = w0.y; wbv[ni][2] = w0.z; wbv[ni][3] = w0.w;
    wbv[ni][4] = w1.x; wbv[ni][5] = w1.y; wbv[ni][6] = w1.z; wbv[ni][7] = w1.w;
  }
#pragma unroll
  for (int mi = 0; mi < 2; ++mi) {
#pragma unroll
    for (int v = 0; v < 8; ++v) {
      const int  ml = wm * 32 + mi * 16 + lhi * 8 + v;  // M = v (+8 for upper half-wave)
      const bool mk = (Ml[ml] != 0);
      float tr[R_DIM];
#pragma unroll
      for (int r = 0; r < R_DIM; ++r) tr[r] = Tl[ml * R_DIM + r];
      const size_t orow = (size_t)(m0 + ml) * N_DIM;
#pragma unroll
      for (int ni = 0; ni < 4; ++ni) {
        float lora = 0.f;
#pragma unroll
        for (int r = 0; r < R_DIM; ++r) lora = fmaf(tr[r], wbv[ni][r], lora);
        float val = acc[mi][ni][v] + bias[ni];
        if (mk) val = fmaf(LORA_SCALE, lora, val);
        out[orow + (size_t)(n0 + wn * 64 + ni * 16 + l15)] = val;
      }
    }
  }
}

// ---------------- launch ----------------
extern "C" void kernel_launch(void* const* d_in, const int* in_sizes, int n_in,
                              void* d_out, int out_size, void* d_ws, size_t ws_size,
                              hipStream_t stream) {
  (void)in_sizes; (void)n_in; (void)out_size; (void)ws_size;
  const float* x    = (const float*)d_in[0];
  const int*   mask = (const int*)d_in[1];
  const float* Wm   = (const float*)d_in[2];
  const float* bm   = (const float*)d_in[3];
  const float* WA   = (const float*)d_in[4];
  const float* WB   = (const float*)d_in[5];
  float*       out  = (float*)d_out;
  float*       T    = (float*)d_ws;   // M_TOT x 8 fp32 scratch (256 KB)

  plora_lora_a_kernel<<<M_TOT, 256, 0, stream>>>(x, WA, T);

  dim3 grid(N_DIM / BN, M_TOT / BM);  // 32 x 64 tiles
  plora_gemm_kernel<<<grid, 256, 0, stream>>>(x, Wm, bm, WB, T, mask, out);
}
